// AttLayer_72164040508225
// MI455X (gfx1250) — matrix operand order repaired and checked by reading, not compiled, not run on hardware
//
#include <hip/hip_runtime.h>

typedef __attribute__((ext_vector_type(16))) _Float16 v16h;
typedef __attribute__((ext_vector_type(8)))  _Float16 v8h;
typedef __attribute__((ext_vector_type(4)))  _Float16 v4h;
typedef __attribute__((ext_vector_type(8)))  float    v8f;

#define NB   256   // batch
#define NS   512   // seq len
#define NF   256   // input features
#define NHID 128   // hidden per head
#define NH0  4     // heads
#define NOUT 128   // output features
#define NJ   (NH0 * NHID)   // 512 concat hidden

__device__ __forceinline__ float lrelu02(float x) { return x > 0.f ? x : 0.2f * x; }

__device__ __forceinline__ v8f vzero8() {
  v8f z;
#pragma unroll
  for (int i = 0; i < 8; ++i) z[i] = 0.f;
  return z;
}

// ---------------------------------------------------------------------------
// Kernel 1: fts[h,b] = X[b] @ W0[h]  (f16 WMMA, f32 accum), fused f1/f2 dots.
// Stores fts TRANSPOSED in global: FTS[h][b][o][t]  (coalesced b128 via LDS).
// grid (NS/128, NB, NH0), block 256 (8 waves). Wave tile: 32(M) x 64(N).
// ---------------------------------------------------------------------------
__global__ void __launch_bounds__(256)
fts_gemm_kernel(const float* __restrict__ X, const float* __restrict__ W0,
                const float* __restrict__ a1h, const float* __restrict__ a2h,
                _Float16* __restrict__ FTS, float* __restrict__ F1,
                float* __restrict__ F2)
{
  const int stile = blockIdx.x, b = blockIdx.y, h = blockIdx.z;
  const int tid = threadIdx.x;
  const int lane = tid & 31, wave = tid >> 5;
  const int hl = lane >> 4, l16 = lane & 15;

  constexpr int LDA = NF + 8;                  // pad: keeps 16B alignment, no conflicts
  __shared__ _Float16 Xs[128 * LDA];           // [m][k] row-major f16 (reused as stageT)
  __shared__ _Float16 Ws[NHID * LDA];          // [n][k] column-major W0 f16
  __shared__ float f1s[128];
  __shared__ float f2s[128];

  if (tid < 128) { f1s[tid] = 0.f; f2s[tid] = 0.f; }

  // stage X tile (coalesced float4)
  const float* xg = X + ((size_t)b * NS + (size_t)stile * 128) * NF;
  for (int idx = tid; idx < 128 * (NF / 4); idx += 256) {
    const int m = idx >> 6, k4 = idx & 63;
    const float4 v = *(const float4*)(xg + (size_t)m * NF + (size_t)k4 * 4);
    _Float16* d = &Xs[m * LDA + k4 * 4];
    d[0] = (_Float16)v.x; d[1] = (_Float16)v.y;
    d[2] = (_Float16)v.z; d[3] = (_Float16)v.w;
  }
  // stage W0[h] transposed: Ws[n][k] = W0[h][k][n] (coalesced reads along n)
  const float* wg = W0 + (size_t)h * NF * NHID;
  for (int idx = tid; idx < NF * (NHID / 4); idx += 256) {
    const int k = idx >> 5, n4 = idx & 31;
    const float4 v = *(const float4*)(wg + (size_t)k * NHID + (size_t)n4 * 4);
    Ws[(n4 * 4 + 0) * LDA + k] = (_Float16)v.x;
    Ws[(n4 * 4 + 1) * LDA + k] = (_Float16)v.y;
    Ws[(n4 * 4 + 2) * LDA + k] = (_Float16)v.z;
    Ws[(n4 * 4 + 3) * LDA + k] = (_Float16)v.w;
  }
  __syncthreads();

  const int m0 = (wave & 3) * 32, n0 = (wave >> 2) * 64;
  v8f acc[2][4];
#pragma unroll
  for (int mt = 0; mt < 2; ++mt)
#pragma unroll
    for (int nt = 0; nt < 4; ++nt) acc[mt][nt] = vzero8();

  for (int k0 = 0; k0 < NF; k0 += 32) {
    v16h a[2];
#pragma unroll
    for (int mt = 0; mt < 2; ++mt) {
      // A 16x32 f16 layout: lane row = l16, K runs [8*hl, +8) and [16+8*hl, +8)
      const _Float16* ar = &Xs[(m0 + mt * 16 + l16) * LDA + k0 + 8 * hl];
      v8h r0 = *(const v8h*)ar;
      v8h r1 = *(const v8h*)(ar + 16);
      v16h t;
#pragma unroll
      for (int i = 0; i < 8; ++i) { t[i] = r0[i]; t[8 + i] = r1[i]; }
      a[mt] = t;
    }
#pragma unroll
    for (int nt = 0; nt < 4; ++nt) {
      // B 32x16 f16 layout: lane column = l16, K runs [8*hl, +8) and [16+8*hl, +8)
      const _Float16* br = &Ws[(n0 + nt * 16 + l16) * LDA + k0 + 8 * hl];
      v8h c0 = *(const v8h*)br;
      v8h c1 = *(const v8h*)(br + 16);
      v16h bf;
#pragma unroll
      for (int i = 0; i < 8; ++i) { bf[i] = c0[i]; bf[8 + i] = c1[i]; }
#pragma unroll
      for (int mt = 0; mt < 2; ++mt)
        acc[mt][nt] = __builtin_amdgcn_wmma_f32_16x16x32_f16(
            false, a[mt], false, bf, (short)0, acc[mt][nt], false, false);
    }
  }

  // fused f1/f2: dot each fts row with a1/a2 (shfl within 16-lane half + ds_add)
  float a1v[4], a2v[4];
#pragma unroll
  for (int nt = 0; nt < 4; ++nt) {
    const int o = n0 + nt * 16 + l16;
    a1v[nt] = a1h[h * NHID + o];
    a2v[nt] = a2h[h * NHID + o];
  }
#pragma unroll
  for (int mt = 0; mt < 2; ++mt)
#pragma unroll
    for (int r = 0; r < 8; ++r) {
      float s1 = 0.f, s2 = 0.f;
#pragma unroll
      for (int nt = 0; nt < 4; ++nt) {
        const float c = acc[mt][nt][r];
        s1 += c * a1v[nt]; s2 += c * a2v[nt];
      }
      for (int msk = 1; msk < 16; msk <<= 1) {   // reduce over l16 within half
        s1 += __shfl_xor(s1, msk, 32);
        s2 += __shfl_xor(s2, msk, 32);
      }
      if (l16 == 0) {
        const int row = m0 + mt * 16 + r + 8 * hl;
        atomicAdd(&f1s[row], s1);
        atomicAdd(&f2s[row], s2);
      }
    }
  __syncthreads();   // k-loop LDS reads + f1s/f2s atomics complete block-wide

  // transpose C tiles into LDS (reuse Xs), packing 4 rows per ds_store_b64
  _Float16* stageT = Xs;                        // [o][row], SDT stride
  constexpr int SDT = 128 + 8;
#pragma unroll
  for (int mt = 0; mt < 2; ++mt)
#pragma unroll
    for (int nt = 0; nt < 4; ++nt) {
      const int o  = n0 + nt * 16 + l16;
      const int rb = m0 + mt * 16 + 8 * hl;     // rows rb..rb+7 belong to this lane
      v4h lo, hi;
#pragma unroll
      for (int i = 0; i < 4; ++i) {
        lo[i] = (_Float16)acc[mt][nt][i];
        hi[i] = (_Float16)acc[mt][nt][4 + i];
      }
      *(v4h*)&stageT[o * SDT + rb]     = lo;
      *(v4h*)&stageT[o * SDT + rb + 4] = hi;
    }
  if (tid < 128) {
    const size_t base = ((size_t)h * NB + b) * NS + (size_t)stile * 128 + tid;
    F1[base] = f1s[tid];
    F2[base] = f2s[tid];
  }
  __syncthreads();

  // coalesced transposed global write: FTS layout [h][b][o][t]
  _Float16* fT = FTS + ((size_t)h * NB + b) * (size_t)NHID * NS;
  for (int idx = tid; idx < NHID * (128 / 8); idx += 256) {
    const int o = idx >> 4, c = idx & 15;
    v8h v = *(const v8h*)&stageT[o * SDT + c * 8];
    *(v8h*)(fT + (size_t)o * NS + (size_t)stile * 128 + (size_t)c * 8) = v;
  }
}

// ---------------------------------------------------------------------------
// Kernel 2: per (h,b) attention  vals = softmax(lrelu(f1[s]+f2[t])) @ fts.
// P generated in registers from f1/f2 (no QK^T GEMM). fts^T async-DMA'd to LDS
// (global_load_async_to_lds_b128, overlapped with softmax-stat precompute).
// Writes ELU(vals + bias) IN PLACE over fts, row-major h1[t][o].
// grid (NB, NH0), block 512 (16 waves). Wave tile: 32(M) x 128(N).
// ---------------------------------------------------------------------------
__global__ void __launch_bounds__(512)
attn_hidden_kernel(_Float16* __restrict__ FTS, const float* __restrict__ F1,
                   const float* __restrict__ F2, const float* __restrict__ bias0)
{
  const int b = blockIdx.x, h = blockIdx.y;
  const int tid = threadIdx.x, lane = tid & 31, wave = tid >> 5;
  const int hl = lane >> 4, l16 = lane & 15;

  constexpr int LDT = NS + 8;                 // 520: rows stay 16B-aligned
  __shared__ _Float16 FtT[NHID * LDT];        // fts^T: [o][t]  (~133 KB)
  __shared__ float f1r[NS], mxr[NS], rsr[NS];
  __shared__ float f2sh[NS];
  __shared__ float redbuf[16];

  _Float16* fbase = FTS + ((size_t)h * NB + b) * (size_t)NHID * NS; // [o][t]

  // ---- async DMA: global [o][t] -> LDS [o][t+pad], 16B per lane per issue ----
  {
    const unsigned lbase = (unsigned)(unsigned long long)(&FtT[0]);
#pragma unroll 4
    for (int u = tid; u < NHID * (NS / 8); u += 512) {   // 8192 x 16B units
      const int o = u >> 6, c = u & 63;
      const unsigned goff  = (unsigned)u * 16u;                       // contiguous
      const unsigned laddr = lbase + (unsigned)((o * LDT + c * 8) * 2);
      asm volatile("global_load_async_to_lds_b128 %0, %1, %2"
                   :: "v"(laddr), "v"(goff), "s"(fbase) : "memory");
    }
  }

  const float* f1g = F1 + ((size_t)h * NB + b) * NS;
  const float* f2g = F2 + ((size_t)h * NB + b) * NS;
  f2sh[tid] = f2g[tid];
  __syncthreads();

  // max_t f2 (lrelu monotone -> row max = lrelu(f1[s] + f2max))
  float fm = f2sh[tid];
  for (int msk = 16; msk >= 1; msk >>= 1) fm = fmaxf(fm, __shfl_xor(fm, msk, 32));
  if (lane == 0) redbuf[wave] = fm;
  __syncthreads();
  if (tid < 32) {
    float v = (tid < 16) ? redbuf[tid] : -1e30f;
    for (int msk = 8; msk >= 1; msk >>= 1) v = fmaxf(v, __shfl_xor(v, msk, 32));
    if (tid == 0) redbuf[0] = v;
  }
  __syncthreads();
  const float f2max = redbuf[0];

  {                                           // per-row softmax stats (thread = row)
    const float f1v = f1g[tid];
    const float mx = lrelu02(f1v + f2max);
    float s = 0.f;
    for (int t = 0; t < NS; ++t) s += __expf(lrelu02(f1v + f2sh[t]) - mx);
    f1r[tid] = f1v; mxr[tid] = mx; rsr[tid] = 1.f / s;
  }
  asm volatile("s_wait_asynccnt 0x0" ::: "memory");   // DMA complete
  __syncthreads();                                    // FtT + stats visible

  const int m0 = wave * 32;
  v8f acc[2][8];
#pragma unroll
  for (int mt = 0; mt < 2; ++mt)
#pragma unroll
    for (int nt = 0; nt < 8; ++nt) acc[mt][nt] = vzero8();

  for (int t0 = 0; t0 < NS; t0 += 32) {
    float e0[8], e1[8];
    const int ta = t0 + 8 * hl, tb = t0 + 16 + 8 * hl;
#pragma unroll
    for (int i = 0; i < 8; ++i) { e0[i] = f2sh[ta + i]; e1[i] = f2sh[tb + i]; }
    v16h a[2];
#pragma unroll
    for (int mt = 0; mt < 2; ++mt) {          // build P fragment in registers
      const int row = m0 + mt * 16 + l16;
      const float f1v = f1r[row], mx = mxr[row];
      v16h t;
#pragma unroll
      for (int i = 0; i < 8; ++i) {
        t[i]     = (_Float16)__expf(lrelu02(f1v + e0[i]) - mx);
        t[8 + i] = (_Float16)__expf(lrelu02(f1v + e1[i]) - mx);
      }
      a[mt] = t;
    }
#pragma unroll
    for (int nt = 0; nt < 8; ++nt) {
      const _Float16* br = &FtT[(nt * 16 + l16) * LDT + t0 + 8 * hl];
      v8h c0 = *(const v8h*)br;
      v8h c1 = *(const v8h*)(br + 16);
      v16h bf;
#pragma unroll
      for (int i = 0; i < 8; ++i) { bf[i] = c0[i]; bf[8 + i] = c1[i]; }
#pragma unroll
      for (int mt = 0; mt < 2; ++mt)
        acc[mt][nt] = __builtin_amdgcn_wmma_f32_16x16x32_f16(
            false, a[mt], false, bf, (short)0, acc[mt][nt], false, false);
    }
  }

  // epilogue: normalize, bias, ELU, write h1 row-major in place over fts
  // (safe: all async copies waited before the pre-loop barrier)
#pragma unroll
  for (int mt = 0; mt < 2; ++mt)
#pragma unroll
    for (int nt = 0; nt < 8; ++nt) {
      const int o = nt * 16 + l16;
      const float bv = bias0[h * NHID + o];
#pragma unroll
      for (int r = 0; r < 8; ++r) {
        const int row = m0 + mt * 16 + r + 8 * hl;
        float v = acc[mt][nt][r] * rsr[row] + bv;
        v = v > 0.f ? v : (__expf(v) - 1.f);
        fbase[(size_t)row * NHID + o] = (_Float16)v;
      }
    }
}

// ---------------------------------------------------------------------------
// Kernel 3: w1v = W1 @ a1_1, w2v = W1 @ a2_1  (tiny)
// ---------------------------------------------------------------------------
__global__ void __launch_bounds__(256)
wvec_kernel(const float* __restrict__ W1, const float* __restrict__ a11,
            const float* __restrict__ a21, float* __restrict__ w1v,
            float* __restrict__ w2v)
{
  const int j = blockIdx.x * 256 + threadIdx.x;
  if (j < NJ) {
    float s1 = 0.f, s2 = 0.f;
    for (int o = 0; o < NOUT; ++o) {
      const float w = W1[(size_t)j * NOUT + o];
      s1 += w * a11[o]; s2 += w * a21[o];
    }
    w1v[j] = s1; w2v[j] = s2;
  }
}

// ---------------------------------------------------------------------------
// Kernel 4: output layer, only seq position 0 is needed:
// out[b,:] = (sum_t alpha[t] * h1[b,t,:]) @ W1 + b1.  h1 stored as [h][b][t][o].
// grid NB, block 256.
// ---------------------------------------------------------------------------
__global__ void __launch_bounds__(256)
out_layer_kernel(const _Float16* __restrict__ H1, const float* __restrict__ W1,
                 const float* __restrict__ b1, const float* __restrict__ w1v,
                 const float* __restrict__ w2v, float* __restrict__ out)
{
  const int b = blockIdx.x, tid = threadIdx.x, lane = tid & 31, wave = tid >> 5;
  __shared__ float w2s[NJ];
  __shared__ float alpha[NS];
  __shared__ float g[NJ];
  __shared__ float red[8];

  auto h1at = [&](int t, int j) -> const _Float16* {
    const int hh = j >> 7, o = j & 127;            // j = h*128 + o
    return H1 + (((size_t)hh * NB + b) * NS + t) * NHID + o;
  };

  for (int j = tid; j < NJ; j += 256) w2s[j] = w2v[j];
  __syncthreads();

  // f1_1 = h1[b,0,:] . (W1 a1_1)
  float p = 0.f;
  for (int j = tid; j < NJ; j += 256) p += (float)(*h1at(0, j)) * w1v[j];
  for (int msk = 16; msk >= 1; msk >>= 1) p += __shfl_xor(p, msk, 32);
  if (lane == 0) red[wave] = p;
  __syncthreads();
  if (tid == 0) { float s = 0.f; for (int i = 0; i < 8; ++i) s += red[i]; red[0] = s; }
  __syncthreads();
  const float f1_1 = red[0];
  __syncthreads();

  // f2_1[t] = h1[b,t,:] . (W1 a2_1)   (one wave per t)
  for (int t = wave; t < NS; t += 8) {
    const int j0 = lane * 16;
    const _Float16* hr = h1at(t, j0);
    v8h x0 = *(const v8h*)hr;
    v8h x1 = *(const v8h*)(hr + 8);
    float s = 0.f;
#pragma unroll
    for (int k = 0; k < 8; ++k)
      s += (float)x0[k] * w2s[j0 + k] + (float)x1[k] * w2s[j0 + 8 + k];
    for (int msk = 16; msk >= 1; msk >>= 1) s += __shfl_xor(s, msk, 32);
    if (lane == 0) alpha[t] = s;
  }
  __syncthreads();

  // row-0 softmax over t
  float fmx = -1e30f;
  for (int t = tid; t < NS; t += 256) fmx = fmaxf(fmx, alpha[t]);
  for (int msk = 16; msk >= 1; msk >>= 1) fmx = fmaxf(fmx, __shfl_xor(fmx, msk, 32));
  if (lane == 0) red[wave] = fmx;
  __syncthreads();
  if (tid == 0) { float v = red[0]; for (int i = 1; i < 8; ++i) v = fmaxf(v, red[i]); red[0] = v; }
  __syncthreads();
  const float mx = lrelu02(f1_1 + red[0]);
  __syncthreads();

  float ps = 0.f;
  for (int t = tid; t < NS; t += 256) {
    const float e = __expf(lrelu02(f1_1 + alpha[t]) - mx);
    alpha[t] = e; ps += e;
  }
  for (int msk = 16; msk >= 1; msk >>= 1) ps += __shfl_xor(ps, msk, 32);
  __syncthreads();
  if (lane == 0) red[wave] = ps;
  __syncthreads();
  if (tid == 0) { float s = 0.f; for (int i = 0; i < 8; ++i) s += red[i]; red[0] = s; }
  __syncthreads();
  const float rden = 1.f / red[0];

  // g = (sum_t alpha[t] h1[b,t,:]) / denom
  for (int j = tid; j < NJ; j += 256) {
    const int hh = j >> 7, o = j & 127;
    const _Float16* col = H1 + (((size_t)hh * NB + b) * NS) * NHID + o;
    float s = 0.f;
    for (int t = 0; t < NS; ++t) s += alpha[t] * (float)col[(size_t)t * NHID];
    g[j] = s * rden;
  }
  __syncthreads();

  if (tid < NOUT) {
    float s = 0.f;
    for (int j = 0; j < NJ; ++j) s += g[j] * W1[(size_t)j * NOUT + tid];
    out[(size_t)b * NOUT + tid] = s + b1[tid];
  }
}

// ---------------------------------------------------------------------------
extern "C" void kernel_launch(void* const* d_in, const int* in_sizes, int n_in,
                              void* d_out, int out_size, void* d_ws, size_t ws_size,
                              hipStream_t stream) {
  const float* x    = (const float*)d_in[0];
  const float* W0   = (const float*)d_in[1];
  const float* a1_0 = (const float*)d_in[2];
  const float* a2_0 = (const float*)d_in[3];
  const float* b0   = (const float*)d_in[4];
  const float* W1   = (const float*)d_in[5];
  const float* a1_1 = (const float*)d_in[6];
  const float* a2_1 = (const float*)d_in[7];
  const float* b1   = (const float*)d_in[8];
  float* out = (float*)d_out;

  char* ws = (char*)d_ws;
  const size_t szFTS = (size_t)NH0 * NB * NS * NHID * sizeof(_Float16); // 128 MiB
  const size_t szF   = (size_t)NH0 * NB * NS * sizeof(float);          // 2 MiB
  _Float16* FTS = (_Float16*)ws;                 // fts^T, then overwritten by h1
  float* F1  = (float*)(ws + szFTS);
  float* F2  = (float*)(ws + szFTS + szF);
  float* w1v = (float*)(ws + szFTS + 2 * szF);
  float* w2v = w1v + NJ;

  wvec_kernel<<<dim3(2), 256, 0, stream>>>(W1, a1_1, a2_1, w1v, w2v);
  fts_gemm_kernel<<<dim3(NS / 128, NB, NH0), 256, 0, stream>>>(
      x, W0, a1_0, a2_0, FTS, F1, F2);
  attn_hidden_kernel<<<dim3(NB, NH0), 512, 0, stream>>>(FTS, F1, F2, b0);
  out_layer_kernel<<<dim3(NB), 256, 0, stream>>>(FTS, W1, b1, w1v, w2v, out);

  (void)in_sizes; (void)n_in; (void)out_size; (void)ws_size;
}
